// SelfAttentionHead_326417515274
// MI455X (gfx1250) — matrix operand order repaired and checked by reading, not compiled, not run on hardware
//
#include <hip/hip_runtime.h>
#include <hip/hip_bf16.h>

// ---------------------------------------------------------------------------
// out = leaky_relu( (xs @ Wv^T + bv) @ Wlin^T + blin )
// (sum over softmax axis == 1 -> attention block is algebraically identity)
// B=16, S=1024, D=1024  =>  M = 16384, N = K = 1024
//
// GEMMs: bf16 WMMA (v_wmma_f32_16x16x32_bf16) fed from LDS, with the global->
// LDS traffic done by the CDNA5 async DMA path (global_load_async_to_lds_b128,
// tracked with ASYNCcnt / s_wait_asynccnt), double-buffered.
// ---------------------------------------------------------------------------

typedef __attribute__((ext_vector_type(16))) __bf16 v16bf;
typedef __attribute__((ext_vector_type(8)))  __bf16 v8bf;
typedef __attribute__((ext_vector_type(4)))  __bf16 v4bf;
typedef __attribute__((ext_vector_type(8)))  float  v8f;
typedef __attribute__((ext_vector_type(4)))  float  v4f;

union FragAB { v16bf v; v8bf h[2]; };

// ---------------------------------------------------------------------------
// f32 -> bf16 conversion (vectorized, 4 elements/thread)
// ---------------------------------------------------------------------------
__global__ __launch_bounds__(256) void cvt_f32_bf16(
    const float* __restrict__ in, __bf16* __restrict__ out, int n)
{
    int i = (blockIdx.x * 256 + threadIdx.x) * 4;
    if (i + 3 < n) {
        v4f x = *(const v4f*)(in + i);
        v4bf y;
        y.x = (__bf16)x.x;  y.y = (__bf16)x.y;
        y.z = (__bf16)x.z;  y.w = (__bf16)x.w;
        *(v4bf*)(out + i) = y;
    }
}

// ---------------------------------------------------------------------------
// NT bf16 GEMM, f32 accumulate:  C[m,n] = sum_k A[m,k] * W[n,k] + bias[n]
//   A : [M,K] row-major bf16,  W : [N,K] row-major bf16
// Block: 256 threads = 8 waves (2 M x 4 N), block tile 64 x 256.
// K-stage 32, LDS double buffer, async global->LDS copies.
// Rows padded to 80B in LDS: bank stride 20 dwords, gcd(20,64)=4 ->
// 16 consecutive rows hit 16 distinct banks (conflict-free ds_load_b128).
// MODE 0: store bf16     MODE 1: bias + leaky_relu, store f32
// ---------------------------------------------------------------------------
#define KSTEP     32
#define A_STRIDE  80                       // 64B row + 16B pad
#define B_STRIDE  80
#define A_BYTES   (64  * A_STRIDE)         // 5120
#define B_BYTES   (256 * B_STRIDE)         // 20480
#define STG_BYTES (A_BYTES + B_BYTES)      // 25600 per buffer (x2 = 50KB LDS)

template<int MODE>
__global__ __launch_bounds__(256) void gemm_bf16_nt(
    const __bf16* __restrict__ A,
    const __bf16* __restrict__ W,
    const float*  __restrict__ bias,
    void* __restrict__ outp,
    int M, int N, int K)
{
    __shared__ __align__(16) char smem[2 * STG_BYTES];

    const int tid  = threadIdx.x;
    const int wave = tid >> 5;
    const int lane = tid & 31;
    const int lm   = lane & 15;     // lane-within-half: M index (A) / N index (B)
    const int kh   = lane >> 4;     // lane half selects K sub-block

    const int wm = wave >> 2;       // 0..1
    const int wn = wave & 3;        // 0..3

    const int m0b = blockIdx.y * 64;
    const int n0b = blockIdx.x * 256;

    // ---- async-copy assignment: per thread 1 A chunk + 4 B chunks (16B each)
    const int crow = tid >> 2;             // 0..63
    const int ccol = (tid & 3) * 16;       // byte offset within 64B k-row
    const uint32_t lds0 = (uint32_t)(uintptr_t)smem;

    const char*    gA = (const char*)A + ((size_t)(m0b + crow) * K) * 2 + ccol;
    const uint32_t lA = (uint32_t)(crow * A_STRIDE + ccol);

    const char* gB[4];
    uint32_t    lB[4];
#pragma unroll
    for (int j = 0; j < 4; ++j) {
        const int row = crow + j * 64;     // 0..255
        gB[j] = (const char*)W + ((size_t)(n0b + row) * K) * 2 + ccol;
        lB[j] = (uint32_t)(A_BYTES + row * B_STRIDE + ccol);
    }

    const int STAGES = K / KSTEP;          // 32

    auto issue = [&](int s) {
        const uint32_t boff = lds0 + (uint32_t)(s & 1) * STG_BYTES;
        const size_t   kb   = (size_t)s * (KSTEP * 2);   // byte advance along K
        asm volatile("global_load_async_to_lds_b128 %0, %1, off"
                     :: "v"(boff + lA), "v"((uint64_t)(uintptr_t)(gA + kb))
                     : "memory");
#pragma unroll
        for (int j = 0; j < 4; ++j)
            asm volatile("global_load_async_to_lds_b128 %0, %1, off"
                         :: "v"(boff + lB[j]), "v"((uint64_t)(uintptr_t)(gB[j] + kb))
                         : "memory");
    };

    v8f acc[2][4];
#pragma unroll
    for (int mt = 0; mt < 2; ++mt)
#pragma unroll
        for (int nt = 0; nt < 4; ++nt)
            acc[mt][nt] = (v8f){0.f, 0.f, 0.f, 0.f, 0.f, 0.f, 0.f, 0.f};

    issue(0);

    for (int s = 0; s < STAGES; ++s) {
        if (s + 1 < STAGES) {
            issue(s + 1);
            // 5 ops in flight for stage s+1; async loads retire in order,
            // so ASYNCcnt<=5 means stage s has fully landed in LDS.
            asm volatile("s_wait_asynccnt 0x5" ::: "memory");
        } else {
            asm volatile("s_wait_asynccnt 0x0" ::: "memory");
        }
        __syncthreads();

        const char* aS = smem + (s & 1) * STG_BYTES;
        const char* bS = aS + A_BYTES;

        FragAB a[2], b[4];
        // A frag 16x32: half 0 -> K {0..7,16..23}, half 1 -> {8..15,24..31}
#pragma unroll
        for (int mt = 0; mt < 2; ++mt) {
            const char* ap = aS + (wm * 32 + mt * 16 + lm) * A_STRIDE + kh * 16;
            a[mt].h[0] = *(const v8bf*)(ap);
            a[mt].h[1] = *(const v8bf*)(ap + 32);
        }
        // B frag 32x16: half 0 -> K {0..7,16..23}, half 1 -> {8..15,24..31}
#pragma unroll
        for (int nt = 0; nt < 4; ++nt) {
            const char* bp = bS + (wn * 64 + nt * 16 + lm) * B_STRIDE + kh * 16;
            b[nt].h[0] = *(const v8bf*)(bp);
            b[nt].h[1] = *(const v8bf*)(bp + 32);
        }

#pragma unroll
        for (int mt = 0; mt < 2; ++mt)
#pragma unroll
            for (int nt = 0; nt < 4; ++nt)
                acc[mt][nt] = __builtin_amdgcn_wmma_f32_16x16x32_bf16(
                    false, a[mt].v, false, b[nt].v,
                    (short)0, acc[mt][nt], false, false);

        __syncthreads();   // readers done before buffer (s&1) is overwritten
    }

    // Epilogue. C/D layout: lanes 0-15 VGPR r -> (M=r, N=lane);
    //                       lanes 16-31 VGPR r -> (M=r+8, N=lane-16)
    const int m0 = m0b + wm * 32;
    const int n0 = n0b + wn * 64;
#pragma unroll
    for (int mt = 0; mt < 2; ++mt) {
#pragma unroll
        for (int nt = 0; nt < 4; ++nt) {
            const int n  = n0 + nt * 16 + lm;
            const float bn = bias[n];
#pragma unroll
            for (int r = 0; r < 8; ++r) {
                const int m = m0 + mt * 16 + kh * 8 + r;
                float c = acc[mt][nt][r] + bn;
                if (MODE == 0) {
                    ((__bf16*)outp)[(size_t)m * N + n] = (__bf16)c;
                } else {
                    ((float*)outp)[(size_t)m * N + n] = (c >= 0.f) ? c : 0.01f * c;
                }
            }
        }
    }
}

// ---------------------------------------------------------------------------
extern "C" void kernel_launch(void* const* d_in, const int* in_sizes, int n_in,
                              void* d_out, int out_size, void* d_ws, size_t ws_size,
                              hipStream_t stream)
{
    (void)in_sizes; (void)n_in; (void)out_size; (void)ws_size;

    const float* xs   = (const float*)d_in[0];
    // d_in[1] mask, d_in[2..5] Wk/bk/Wq/bq: dead (sum of softmax == 1)
    const float* Wv   = (const float*)d_in[6];
    const float* bv   = (const float*)d_in[7];
    const float* Wlin = (const float*)d_in[8];
    const float* blin = (const float*)d_in[9];

    const int M = 16 * 1024;          // B*S
    const int D = 1024;
    const size_t MD = (size_t)M * D;  // 16,777,216
    const size_t DD = (size_t)D * D;  // 1,048,576

    char* ws = (char*)d_ws;
    __bf16* Xbf  = (__bf16*)(ws);
    __bf16* Vbf  = (__bf16*)(ws + MD * 2);
    __bf16* Wvbf = (__bf16*)(ws + MD * 4);
    __bf16* Wlbf = (__bf16*)(ws + MD * 4 + DD * 2);

    // 1) convert operands to bf16
    cvt_f32_bf16<<<(int)(MD / 1024), 256, 0, stream>>>(xs, Xbf, (int)MD);
    cvt_f32_bf16<<<(int)(DD / 1024), 256, 0, stream>>>(Wv, Wvbf, (int)DD);
    cvt_f32_bf16<<<(int)(DD / 1024), 256, 0, stream>>>(Wlin, Wlbf, (int)DD);

    // 2) V = xs @ Wv^T + bv              (store bf16)
    // 3) out = leaky(V @ Wlin^T + blin)  (store f32)
    dim3 grid(D / 256, M / 64);       // (4, 256)
    gemm_bf16_nt<0><<<grid, 256, 0, stream>>>(Xbf, Wvbf, bv, (void*)Vbf, M, D, D);
    gemm_bf16_nt<1><<<grid, 256, 0, stream>>>(Vbf, Wlbf, blin, d_out, M, D, D);
}
